// AGIFORMER_67499706024581
// MI455X (gfx1250) — compile-verified, matrix operand order checked by reading, not run on hardware
//
#include <hip/hip_runtime.h>
#include <hip/hip_bf16.h>
#include <math.h>

typedef __bf16 bf16;
typedef __attribute__((ext_vector_type(16))) __bf16 v16bf;
typedef __attribute__((ext_vector_type(8)))  float  v8f;

#define DD 768
#define FF 3072
#define SS 1024
#define BB 4
#define HH 12
#define LL 4
#define EE 4
#define VV 256
#define TT (BB*SS)
#define DHd 64

// ---------------------------------------------------------------- helpers
// gelu(tanh approx): 0.5v(1+tanh(u)) == v*sigmoid(2u); branch-free fast path
__device__ __forceinline__ float gelu_tanh(float v) {
    float u = 1.5957691216057308f * (v + 0.044715f * v * v * v);  // 2*sqrt(2/pi)
    return v * (1.0f / (1.0f + __expf(-u)));
}

// WMMA fragment loaders per CDNA5 ISA 7.12.2 (16-bit A/B 16x32 layout):
// lane l: row/col = l&15; K elements at 8*(l>>4)+0..7 and 16+8*(l>>4)+0..7.
// Caller passes p already offset by row*stride + 8*(l>>4).
__device__ __forceinline__ v16bf frag_bf16(const bf16* p) {
    v16bf f;
#pragma unroll
    for (int i = 0; i < 8; ++i) { f[i] = p[i]; f[i + 8] = p[16 + i]; }
    return f;
}
__device__ __forceinline__ v16bf frag_f32(const float* p) {
    v16bf f;
#pragma unroll
    for (int i = 0; i < 8; ++i) { f[i] = (bf16)p[i]; f[i + 8] = (bf16)p[16 + i]; }
    return f;
}
__device__ __forceinline__ v8f wmma_bf16(v16bf a, v16bf b, v8f c) {
    return __builtin_amdgcn_wmma_f32_16x16x32_bf16(false, a, false, b, (short)0, c,
                                                   false, false);
}

// CDNA5 async copy: 16B global -> LDS, tracked by ASYNCcnt.
// LDS byte address = low 32 bits of flat LDS pointer (aperture rule).
__device__ __forceinline__ void async_copy16(void* lds_ptr, const void* gptr) {
    unsigned lds_off = (unsigned)(uintptr_t)lds_ptr;
    unsigned long long ga = (unsigned long long)(uintptr_t)gptr;
    asm volatile("global_load_async_to_lds_b128 %0, %1, off"
                 :: "v"(lds_off), "v"(ga) : "memory");
}
__device__ __forceinline__ void wait_async0() {
    asm volatile("s_wait_asynccnt 0x0" ::: "memory");
}

// ---------------------------------------------------------------- pack conv_w
// conv_w [D(out), D(in), 3] 'OIH'  ->  Wc fp32 [3*D (h,in), D (out)] row-major
__global__ __launch_bounds__(256) void pack_convw_kernel(const float* __restrict__ cw,
                                                         float* __restrict__ Wc) {
    int idx = blockIdx.x * 256 + threadIdx.x;          // over 3*D*D
    if (idx >= 3 * DD * DD) return;
    int o   = idx % DD;
    int row = idx / DD;                                // h*DD + i
    int hh  = row / DD;
    int i   = row % DD;
    Wc[idx] = cw[(o * DD + i) * 3 + hh];
}

// ---------------------------------------------------------------- embed + shifted concat
__global__ __launch_bounds__(256) void embed_kernel(const int* __restrict__ text,
                                                    const float* __restrict__ emb,
                                                    float* __restrict__ x,
                                                    bf16* __restrict__ Ac) {
    int t = blockIdx.x;                  // 0..T-1
    int b = t >> 10, s = t & (SS - 1);
    int tok = text[t];
#pragma unroll
    for (int it = 0; it < 3; ++it) {
        int d = threadIdx.x + it * 256;  // D = 768 = 3*256
        float v = emb[(size_t)tok * DD + d];
        x[(size_t)t * DD + d] = v;
#pragma unroll
        for (int hh = 0; hh < 3; ++hh) {
            int ss = s + hh - 1;
            bf16 av = (bf16)0.0f;
            if (ss >= 0 && ss < SS)
                av = (bf16)emb[(size_t)text[b * SS + ss] * DD + d];
            Ac[(size_t)t * (3 * DD) + hh * DD + d] = av;
        }
    }
}

// ---------------------------------------------------------------- generic WMMA GEMM
// C = epilogue(A[MxK]bf16 @ B[KxN]fp32 + bias[N])
// Block tile 128x256 (8 waves, each 64x64 = 4x4 WMMA frags -> 16 wmma / K-tile).
// flags: 1=gelu  2=residual add (Cf +=)  4=gate scale (gate[m*gstride])  8=bf16 out
#define GF_GELU  1
#define GF_RES   2
#define GF_GATE  4
#define GF_BF16  8

__global__ __launch_bounds__(256) void gemm_kernel(const bf16* __restrict__ A,
                                                   const float* __restrict__ B,
                                                   const float* __restrict__ bias,
                                                   float* Cf, bf16* Cb,
                                                   const float* __restrict__ gate,
                                                   int gstride,
                                                   int M, int N, int K, int flags) {
    // double-buffered LDS: A via async-to-LDS, B staged transposed through regs
    __shared__ bf16 As[2][128 * 40];   // [row][k], stride 40 halfwords
    __shared__ bf16 Bs[2][256 * 40];   // [n][k]

    const int tid  = threadIdx.x;
    const int lane = tid & 31, w = tid >> 5;
    const int r16  = lane & 15, half = lane >> 4;
    const int wm   = (w & 1) * 64;        // wave row offset in 128
    const int wn   = (w >> 1) * 64;       // wave col offset in 256
    const int bm   = blockIdx.y * 128, bn = blockIdx.x * 256;

    v8f c[4][4] = {};

    auto stage_A_async = [&](int kt, int buf) {
#pragma unroll
        for (int it = 0; it < 2; ++it) {
            int idx = tid + it * 256;
            int row = idx >> 2, ch = (idx & 3) * 8;
            async_copy16(&As[buf][row * 40 + ch],
                         A + (size_t)(bm + row) * K + kt + ch);
        }
    };

    // B staging: 32x256 fp32 per tile = 512 slots of (2 k-rows x 8 n).
    // Each slot packs (k,k+1) bf16 pairs -> 8 ds_store_b32 into Bs[n][k].
    float4 br[2][4];
    auto load_B_regs = [&](int kt) {
#pragma unroll
        for (int it = 0; it < 2; ++it) {
            int slot = tid + it * 256;
            int kk = (slot >> 5) * 2, nn = (slot & 31) * 8;
            const float* gp = B + (size_t)(kt + kk) * N + bn + nn;
            br[it][0] = *(const float4*)gp;
            br[it][1] = *(const float4*)(gp + 4);
            br[it][2] = *(const float4*)(gp + N);
            br[it][3] = *(const float4*)(gp + N + 4);
        }
    };
    auto store_B_lds = [&](int buf) {
#pragma unroll
        for (int it = 0; it < 2; ++it) {
            int slot = tid + it * 256;
            int kk = (slot >> 5) * 2, nn = (slot & 31) * 8;
            const float* lo = (const float*)&br[it][0];   // row k,   8 n
            const float* hi = (const float*)&br[it][2];   // row k+1, 8 n
#pragma unroll
            for (int i = 0; i < 8; ++i) {
                union { bf16 h[2]; unsigned u; } pk;
                pk.h[0] = (bf16)lo[i];
                pk.h[1] = (bf16)hi[i];
                *(unsigned*)&Bs[buf][(nn + i) * 40 + kk] = pk.u;
            }
        }
    };

    const int nk = K >> 5;
    stage_A_async(0, 0);
    load_B_regs(0);

    for (int t = 0; t < nk; ++t) {
        const int buf = t & 1;
        store_B_lds(buf);
        wait_async0();           // our wave's async A(t) landed in LDS
        __syncthreads();         // all waves' A(t)/B(t) visible

        if (t + 1 < nk) {        // prefetch next tile during compute
            stage_A_async((t + 1) << 5, buf ^ 1);
            load_B_regs((t + 1) << 5);
        }

        v16bf af[4], bq[4];
#pragma unroll
        for (int mi = 0; mi < 4; ++mi)
            af[mi] = frag_bf16(&As[buf][(wm + mi * 16 + r16) * 40 + half * 8]);
#pragma unroll
        for (int ni = 0; ni < 4; ++ni)
            bq[ni] = frag_bf16(&Bs[buf][(wn + ni * 16 + r16) * 40 + half * 8]);
#pragma unroll
        for (int mi = 0; mi < 4; ++mi)
#pragma unroll
            for (int ni = 0; ni < 4; ++ni)
                c[mi][ni] = wmma_bf16(af[mi], bq[ni], c[mi][ni]);
        __syncthreads();         // done reading buf before it is refilled
    }

    // epilogue: C lane mapping M = j + 8*half, N = r16
#pragma unroll
    for (int mi = 0; mi < 4; ++mi) {
#pragma unroll
        for (int ni = 0; ni < 4; ++ni) {
            int n = bn + wn + ni * 16 + r16;
            float bv = bias[n];
#pragma unroll
            for (int j = 0; j < 8; ++j) {
                int m = bm + wm + mi * 16 + half * 8 + j;
                float v = c[mi][ni][j] + bv;
                if (flags & GF_GELU) v = gelu_tanh(v);
                if (flags & GF_GATE) v *= gate[(size_t)m * gstride];
                size_t idx = (size_t)m * N + n;
                if (flags & GF_RES) v += Cf[idx];
                if (flags & GF_BF16) Cb[idx] = (bf16)v;
                else                 Cf[idx] = v;
            }
        }
    }
}

// ---------------------------------------------------------------- attention
// One workgroup per (q-block of 64 rows, head, batch). Full 64x1024 fp32 score
// block lives in dynamic LDS (WGP has 320KB).
#define ATTN_LDS (64 * 1024 * 4 + 64 * 4 + 64 * 40 * 2)

__global__ __launch_bounds__(256) void attn_kernel(const bf16* __restrict__ q,
                                                   const bf16* __restrict__ k,
                                                   const bf16* __restrict__ v,
                                                   bf16* __restrict__ o) {
    extern __shared__ char smem[];
    float* scores = (float*)smem;                           // [64][1024]
    float* rsum   = (float*)(smem + 64 * 1024 * 4);         // [64]
    bf16*  Vt     = (bf16*)(smem + 64 * 1024 * 4 + 64 * 4); // [64][40]

    const int qb = blockIdx.x, hd = blockIdx.y, bb = blockIdx.z;
    const int tid = threadIdx.x, lane = tid & 31, w = tid >> 5;
    const int r16 = lane & 15, half = lane >> 4;
    const float scale = 0.125f;  // 1/sqrt(64)

    // ---- phase 1: S = scale * Q K^T  (waves: 4 q-subtiles x 2 kv-halves)
    {
        const int qs = w & 3, kvh = w >> 2;
        int s_q = qb * 64 + qs * 16 + r16;
        size_t qrow = ((size_t)(bb * SS + s_q)) * DD + hd * DHd;
        v16bf aq0 = frag_bf16(q + qrow + half * 8);
        v16bf aq1 = frag_bf16(q + qrow + 32 + half * 8);
        for (int nt = 0; nt < 32; ++nt) {
            int s_k = kvh * 512 + nt * 16 + r16;
            size_t krow = ((size_t)(bb * SS + s_k)) * DD + hd * DHd;
            v16bf bk0 = frag_bf16(k + krow + half * 8);
            v16bf bk1 = frag_bf16(k + krow + 32 + half * 8);
            v8f c = {};
            c = wmma_bf16(aq0, bk0, c);
            c = wmma_bf16(aq1, bk1, c);
            int col = kvh * 512 + nt * 16 + r16;
#pragma unroll
            for (int j = 0; j < 8; ++j)
                scores[(qs * 16 + half * 8 + j) * 1024 + col] = c[j] * scale;
        }
    }
    __syncthreads();

    // ---- phase 2: row softmax (store unnormalized exp, keep row sums)
    for (int rr = 0; rr < 8; ++rr) {
        int r = w * 8 + rr;
        float* row = scores + r * 1024;
        float m = -1e30f;
        for (int i = lane; i < 1024; i += 32) m = fmaxf(m, row[i]);
#pragma unroll
        for (int off = 16; off; off >>= 1) m = fmaxf(m, __shfl_xor(m, off, 32));
        float s = 0.f;
        for (int i = lane; i < 1024; i += 32) {
            float e = __expf(row[i] - m);
            row[i] = e;
            s += e;
        }
#pragma unroll
        for (int off = 16; off; off >>= 1) s += __shfl_xor(s, off, 32);
        if (lane == 0) rsum[r] = s;
    }
    __syncthreads();

    // ---- phase 3: O = P V  (wave: m-tile = w&3, n-tiles {w>>2, (w>>2)+2})
    const int mi = w & 3, ns = w >> 2;
    v8f c0 = {}, c1 = {};
    for (int kt = 0; kt < 32; ++kt) {
        {   // stage V tile [32 kv][64 dh] -> Vt[dh][kv]
            int krow = tid >> 3;
            int ch   = (tid & 7) * 8;
            const bf16* gp = v + ((size_t)(bb * SS + kt * 32 + krow)) * DD + hd * DHd + ch;
            alignas(16) bf16 tmp[8];
            *(uint4*)tmp = *(const uint4*)gp;
#pragma unroll
            for (int i = 0; i < 8; ++i) Vt[(ch + i) * 40 + krow] = tmp[i];
        }
        __syncthreads();
        v16bf ap = frag_f32(scores + (mi * 16 + r16) * 1024 + kt * 32 + half * 8);
        v16bf b0 = frag_bf16(&Vt[(ns * 16 + r16) * 40 + half * 8]);
        v16bf b1 = frag_bf16(&Vt[((ns + 2) * 16 + r16) * 40 + half * 8]);
        c0 = wmma_bf16(ap, b0, c0);
        c1 = wmma_bf16(ap, b1, c1);
        __syncthreads();
    }
#pragma unroll
    for (int j = 0; j < 8; ++j) {
        int rl = mi * 16 + half * 8 + j;
        float inv = 1.0f / rsum[rl];
        int s_o = qb * 64 + rl;
        size_t orow = ((size_t)(bb * SS + s_o)) * DD + hd * DHd;
        o[orow + ns * 16 + r16]       = (bf16)(c0[j] * inv);
        o[orow + (ns + 2) * 16 + r16] = (bf16)(c1[j] * inv);
    }
}

// ---------------------------------------------------------------- layernorm (fp32 in, bf16 out)
__global__ __launch_bounds__(256) void ln_kernel(const float* __restrict__ x,
                                                 const float* __restrict__ g,
                                                 const float* __restrict__ b,
                                                 bf16* __restrict__ out) {
    int t = blockIdx.x, tid = threadIdx.x;
    const float* xr = x + (size_t)t * DD;
    float xv[3], s1 = 0.f, s2 = 0.f;
#pragma unroll
    for (int i = 0; i < 3; ++i) {
        float vv = xr[tid + i * 256];
        xv[i] = vv; s1 += vv; s2 += vv * vv;
    }
#pragma unroll
    for (int off = 16; off; off >>= 1) {
        s1 += __shfl_xor(s1, off, 32);
        s2 += __shfl_xor(s2, off, 32);
    }
    __shared__ float a1[8], a2[8];
    int w = tid >> 5, lane = tid & 31;
    if (lane == 0) { a1[w] = s1; a2[w] = s2; }
    __syncthreads();
    if (w == 0) {
        float t1 = (lane < 8) ? a1[lane] : 0.f;
        float t2 = (lane < 8) ? a2[lane] : 0.f;
#pragma unroll
        for (int off = 4; off; off >>= 1) {
            t1 += __shfl_xor(t1, off, 32);
            t2 += __shfl_xor(t2, off, 32);
        }
        if (lane == 0) { a1[0] = t1; a2[0] = t2; }
    }
    __syncthreads();
    float mean = a1[0] * (1.0f / DD);
    float var  = a2[0] * (1.0f / DD) - mean * mean;
    float inv  = rsqrtf(var + 1e-5f);
#pragma unroll
    for (int i = 0; i < 3; ++i) {
        int d = tid + i * 256;
        out[(size_t)t * DD + d] = (bf16)((xv[i] - mean) * inv * g[d] + b[d]);
    }
}

// ---------------------------------------------------------------- router: softmax + top2 + renorm
__global__ __launch_bounds__(256) void router_kernel(const bf16* __restrict__ h,
                                                     const float* __restrict__ rw,
                                                     const float* __restrict__ rb,
                                                     float* __restrict__ gate) {
    int t = blockIdx.x * 256 + threadIdx.x;
    if (t >= TT) return;
    const bf16* hr = h + (size_t)t * DD;
    float acc[EE] = {rb[0], rb[1], rb[2], rb[3]};
    for (int d = 0; d < DD; ++d) {
        float hv = (float)hr[d];
        const float* wr = rw + d * EE;
#pragma unroll
        for (int e = 0; e < EE; ++e) acc[e] += hv * wr[e];
    }
    float m = fmaxf(fmaxf(acc[0], acc[1]), fmaxf(acc[2], acc[3]));
    float p[EE], s = 0.f;
#pragma unroll
    for (int e = 0; e < EE; ++e) { p[e] = __expf(acc[e] - m); s += p[e]; }
#pragma unroll
    for (int e = 0; e < EE; ++e) p[e] /= s;
    int i1 = 0;
    for (int e = 1; e < EE; ++e) if (p[e] > p[i1]) i1 = e;        // first max
    int i2 = -1;
    for (int e = 0; e < EE; ++e)
        if (e != i1 && (i2 < 0 || p[e] > p[i2])) i2 = e;          // second (first on ties)
    float s2 = p[i1] + p[i2];
    float* gr = gate + (size_t)t * EE;
#pragma unroll
    for (int e = 0; e < EE; ++e)
        gr[e] = (e == i1) ? p[i1] / s2 : (e == i2) ? p[i2] / s2 : 0.f;
}

// ---------------------------------------------------------------- launcher
extern "C" void kernel_launch(void* const* d_in, const int* in_sizes, int n_in,
                              void* d_out, int out_size, void* d_ws, size_t ws_size,
                              hipStream_t stream) {
    (void)in_sizes; (void)n_in; (void)out_size; (void)ws_size;

    const int*   text   = (const int*)  d_in[0];
    const float* emb    = (const float*)d_in[1];
    const float* conv_w = (const float*)d_in[2];
    const float* conv_b = (const float*)d_in[3];
    const float* Wq = (const float*)d_in[4],  *bq = (const float*)d_in[5];
    const float* Wk = (const float*)d_in[6],  *bk = (const float*)d_in[7];
    const float* Wv = (const float*)d_in[8],  *bv = (const float*)d_in[9];
    const float* Wo = (const float*)d_in[10], *bo = (const float*)d_in[11];
    const float* rw = (const float*)d_in[12], *rb = (const float*)d_in[13];
    const float* ew1 = (const float*)d_in[14], *eb1 = (const float*)d_in[15];
    const float* ew2 = (const float*)d_in[16], *eb2 = (const float*)d_in[17];
    const float* ln1g = (const float*)d_in[18], *ln1b = (const float*)d_in[19];
    const float* ln2g = (const float*)d_in[20], *ln2b = (const float*)d_in[21];
    const float* lnfg = (const float*)d_in[22], *lnfb = (const float*)d_in[23];
    const float* outw = (const float*)d_in[24], *outb = (const float*)d_in[25];

    char* ws = (char*)d_ws;
    size_t off = 0;
    auto carve = [&](size_t bytes) {
        void* p = ws + off;
        off += (bytes + 255) & ~(size_t)255;
        return p;
    };
    float* x    = (float*)carve((size_t)TT * DD * 4);
    bf16*  h    = (bf16*) carve((size_t)TT * DD * 2);
    bf16*  qb   = (bf16*) carve((size_t)TT * DD * 2);
    bf16*  kb   = (bf16*) carve((size_t)TT * DD * 2);
    bf16*  vb   = (bf16*) carve((size_t)TT * DD * 2);
    bf16*  ob   = (bf16*) carve((size_t)TT * DD * 2);
    float* gate = (float*)carve((size_t)TT * EE * 4);
    // union: {Ac bf16 [T,3D] + Wc fp32 [3D,D]} before conv GEMM; h1 bf16 [T,F] after
    char*  un   = (char*) carve(25952256);
    bf16*  Ac   = (bf16*)un;
    float* Wc   = (float*)(un + (size_t)TT * 3 * DD * 2);
    bf16*  h1   = (bf16*)un;

    hipFuncSetAttribute(reinterpret_cast<const void*>(&attn_kernel),
                        hipFuncAttributeMaxDynamicSharedMemorySize, ATTN_LDS);

    dim3 blk(256);

    // tokenizer: embed + shifted-concat, conv-as-GEMM with gelu+residual
    pack_convw_kernel<<<(3 * DD * DD + 255) / 256, blk, 0, stream>>>(conv_w, Wc);
    embed_kernel<<<TT, blk, 0, stream>>>(text, emb, x, Ac);
    gemm_kernel<<<dim3(DD / 256, TT / 128), blk, 0, stream>>>(
        Ac, Wc, conv_b, x, nullptr, nullptr, 0, TT, DD, 3 * DD, GF_GELU | GF_RES);

    for (int i = 0; i < LL; ++i) {
        ln_kernel<<<TT, blk, 0, stream>>>(x, ln1g + i * DD, ln1b + i * DD, h);
        gemm_kernel<<<dim3(DD / 256, TT / 128), blk, 0, stream>>>(
            h, Wq + (size_t)i * DD * DD, bq + i * DD, nullptr, qb, nullptr, 0,
            TT, DD, DD, GF_BF16);
        gemm_kernel<<<dim3(DD / 256, TT / 128), blk, 0, stream>>>(
            h, Wk + (size_t)i * DD * DD, bk + i * DD, nullptr, kb, nullptr, 0,
            TT, DD, DD, GF_BF16);
        gemm_kernel<<<dim3(DD / 256, TT / 128), blk, 0, stream>>>(
            h, Wv + (size_t)i * DD * DD, bv + i * DD, nullptr, vb, nullptr, 0,
            TT, DD, DD, GF_BF16);

        attn_kernel<<<dim3(SS / 64, HH, BB), blk, ATTN_LDS, stream>>>(qb, kb, vb, ob);

        gemm_kernel<<<dim3(DD / 256, TT / 128), blk, 0, stream>>>(
            ob, Wo + (size_t)i * DD * DD, bo + i * DD, x, nullptr, nullptr, 0,
            TT, DD, DD, GF_RES);

        ln_kernel<<<TT, blk, 0, stream>>>(x, ln2g + i * DD, ln2b + i * DD, h);
        router_kernel<<<TT / 256, blk, 0, stream>>>(h, rw + (size_t)i * DD * EE,
                                                    rb + i * EE, gate);
        for (int e = 0; e < EE; ++e) {
            const float* w1 = ew1 + ((size_t)(i * EE + e)) * DD * FF;
            const float* b1 = eb1 + (size_t)(i * EE + e) * FF;
            const float* w2 = ew2 + ((size_t)(i * EE + e)) * FF * DD;
            const float* b2 = eb2 + (size_t)(i * EE + e) * DD;
            gemm_kernel<<<dim3(FF / 256, TT / 128), blk, 0, stream>>>(
                h, w1, b1, nullptr, h1, nullptr, 0, TT, FF, DD, GF_GELU | GF_BF16);
            gemm_kernel<<<dim3(DD / 256, TT / 128), blk, 0, stream>>>(
                h1, w2, b2, x, nullptr, gate + e, EE, TT, DD, FF, GF_RES | GF_GATE);
        }
    }

    ln_kernel<<<TT, blk, 0, stream>>>(x, lnfg, lnfb, h);
    gemm_kernel<<<dim3(VV / 256, TT / 128), blk, 0, stream>>>(
        h, outw, outb, (float*)d_out, nullptr, nullptr, 0, TT, VV, DD, 0);
}
